// NN_28784870818547
// MI455X (gfx1250) — matrix operand order repaired and checked by reading, not compiled, not run on hardware
//
#include <hip/hip_runtime.h>
#include <hip/hip_bf16.h>
#include <math.h>

// CDNA5 / gfx1250: wave32, WMMA f32_16x16x32_f16
typedef __attribute__((ext_vector_type(16))) _Float16 v16h;
typedef __attribute__((ext_vector_type(4)))  _Float16 v4h;
typedef __attribute__((ext_vector_type(8)))  float    v8f;

#define HDIM   100          // LSTM hidden size
#define NG     400          // 4*H gate columns
#define HPAD   128          // K padded to 4 WMMA k-steps of 32
#define MB     16           // batch rows per workgroup (= WMMA M tile)
#define TSTEPS 1024
#define NWAVES 25           // one wave per 16-wide N tile (25*16 = 400)
#define NTHREADS (NWAVES * 32)

__device__ __forceinline__ float softplusf_(float x) { return log1pf(expf(x)); }

// Branch-free activations: gfx1250 has a hardware v_tanh_f32 TRANS op.
#if defined(__has_builtin)
#if __has_builtin(__builtin_amdgcn_tanhf)
#define HAVE_HW_TANH 1
#endif
#endif
__device__ __forceinline__ float fast_tanh(float x) {
#ifdef HAVE_HW_TANH
    return __builtin_amdgcn_tanhf(x);
#else
    float xc = fminf(fmaxf(x, -15.0f), 15.0f);
    float e  = __expf(-2.0f * xc);
    return (1.0f - e) / (1.0f + e);
#endif
}
// sigmoid(x) = 0.5 + 0.5*tanh(x/2): 1 TRANS + 2 VALU, no precise-division sequence
__device__ __forceinline__ float fast_sigmoid(float x) {
    return fmaf(0.5f, fast_tanh(0.5f * x), 0.5f);
}

// 16-bit A fragment (16x32): per-lane K offset inside a 32-wide k-step (ISA 7.12.2)
__device__ __forceinline__ int a_koff(int e, int half) {
    int v = e >> 1, p = e & 1;
    return 2 * (v & 3) + p + ((v & 4) ? 16 : 0) + (half ? 8 : 0);
}
// 16-bit B fragment (32x16): per-lane K offset (ISA 7.12.4 pattern)
__device__ __forceinline__ int b_koff(int e, int half) {
    int v = e >> 1, p = e & 1;
    return 2 * (v & 3) + p + ((v & 4) ? 16 : 0) + (half ? 8 : 0);
}

__device__ __forceinline__ v16h load_a_frag(const _Float16* hbuf, int ks, int row, int half) {
    v16h a;
    const int kb = 32 * ks;
    #pragma unroll
    for (int e = 0; e < 16; ++e)
        a[e] = hbuf[row * HPAD + kb + a_koff(e, half)];
    return a;
}

// k-step 3 of every weight matrix has only k=96..99 nonzero; in the B layout those
// are elements e=0..3 of the half=0 lanes. Keep 2 VGPRs, expand at use.
__device__ __forceinline__ v16h expand_tail(v4h c, int half) {
    v16h f;
    #pragma unroll
    for (int e = 0; e < 16; ++e) f[e] = (_Float16)0.0f;
    #pragma unroll
    for (int e = 0; e < 4; ++e)  f[e] = half ? (_Float16)0.0f : c[e];
    return f;
}

__device__ __forceinline__ v8f wmma_acc(v16h a, v16h b, v8f c) {
    // (neg_a, A, neg_b, B, c_mod, C, reuse_a, reuse_b)
    return __builtin_amdgcn_wmma_f32_16x16x32_f16(false, a, false, b, (short)0, c, false, false);
}

__global__ __launch_bounds__(NTHREADS)
void lstm3_fused(const float* __restrict__ x,
                 const float* __restrict__ wihmu,  const float* __restrict__ wihrho,
                 const float* __restrict__ epsih,
                 const float* __restrict__ whhmu,  const float* __restrict__ whhrho,
                 const float* __restrict__ epshh,
                 const float* __restrict__ bmu,    const float* __restrict__ brho,
                 const float* __restrict__ epsb,
                 const float* __restrict__ l4wih,  const float* __restrict__ l4whh,
                 const float* __restrict__ l4b,
                 const float* __restrict__ linw,   const float* __restrict__ linb,
                 float* __restrict__ out)
{
    __shared__ _Float16 sh[3][MB][HPAD];   // h1/h2/h3, f16, zero-padded K (12 KB)
    __shared__ float    sg[MB][NG];        // gate staging (25.6 KB)
    __shared__ float    sx[MB];            // x_t for this batch chunk

    const int tid   = threadIdx.x;
    const int lane  = tid & 31;
    const int half  = (lane >> 4) & 1;     // lanes 16-31 -> second half of fragment
    const int lrow  = lane & 15;
    const int wv    = tid >> 5;            // wave id 0..24 == N tile id
    const int nCol  = wv * 16 + lrow;      // this lane's gate column
    const int bBase = blockIdx.x * MB;

    // ---------------- one-time init ----------------
    for (int i = tid; i < 3 * MB * HPAD; i += NTHREADS)
        (&sh[0][0][0])[i] = (_Float16)0.0f;                  // h0 = 0, plus K padding

    // per-lane loop-invariant scalars (this lane's gate column)
    const float b1v = bmu[nCol] + softplusf_(brho[nCol]) * epsb[nCol];
    const float b4v = l4b[nCol];
    const float wiv = wihmu[nCol] + softplusf_(wihrho[nCol]) * epsih[nCol]; // W_ih is (1,4H)

    // Loop-invariant B fragments in VGPRs: 3 matrices * (3 full + 1 compact tail)
    // = 3*26 = 78 persistent VGPRs/thread.
    v16h bW1[3], bWi4[3], bWh4[3];
    v4h  cW1,    cWi4,    cWh4;
    #pragma unroll
    for (int ks = 0; ks < 3; ++ks) {        // k = 0..95: all rows valid, no guard
        v16h f1, fi, fh;
        #pragma unroll
        for (int e = 0; e < 16; ++e) {
            const int k   = 32 * ks + b_koff(e, half);
            const int idx = k * NG + nCol;
            f1[e] = (_Float16)(whhmu[idx] + softplusf_(whhrho[idx]) * epshh[idx]);
            fi[e] = (_Float16)l4wih[idx];
            fh[e] = (_Float16)l4whh[idx];
        }
        bW1[ks] = f1; bWi4[ks] = fi; bWh4[ks] = fh;
    }
    #pragma unroll
    for (int e = 0; e < 4; ++e) {           // tail rows k = 96..99 (e -> k = 96+e)
        const int idx = (96 + e) * NG + nCol;
        cW1[e]  = (_Float16)(whhmu[idx] + softplusf_(whhrho[idx]) * epshh[idx]);
        cWi4[e] = (_Float16)l4wih[idx];
        cWh4[e] = (_Float16)l4whh[idx];
    }

    // cell state: 1600 (b,h) pairs / 800 threads = exactly 2 per thread per layer
    float c1[2] = {0.0f, 0.0f}, c2[2] = {0.0f, 0.0f}, c3[2] = {0.0f, 0.0f};

    // ---------------- sequential scan ----------------
    #pragma unroll 1
    for (int t = 0; t < TSTEPS; ++t) {
        if (tid < MB) sx[tid] = x[(bBase + tid) * TSTEPS + t];
        __syncthreads();   // init/prev-step writes visible; gates buffer free

        // ---- layer 1: gates = h1 @ Whh1 + x_t * Wih1 + b1
        {
            v8f acc = {};
            #pragma unroll
            for (int ks = 0; ks < 3; ++ks)
                acc = wmma_acc(load_a_frag(&sh[0][0][0], ks, lrow, half), bW1[ks], acc);
            acc = wmma_acc(load_a_frag(&sh[0][0][0], 3, lrow, half), expand_tail(cW1, half), acc);
            #pragma unroll
            for (int r = 0; r < 8; ++r) {
                const int m = r + (half ? 8 : 0);
                sg[m][nCol] = acc[r] + b1v + sx[m] * wiv;
            }
        }
        __syncthreads();
        #pragma unroll
        for (int q = 0; q < 2; ++q) {
            const int p = tid + q * NTHREADS;
            const int b = p / HDIM, hh = p % HDIM;
            const float ig = fast_sigmoid(sg[b][hh]);
            const float fg = fast_sigmoid(sg[b][hh + HDIM]);
            const float gg = fast_tanh   (sg[b][hh + 2 * HDIM]);
            const float og = fast_sigmoid(sg[b][hh + 3 * HDIM]);
            c1[q] = fg * c1[q] + ig * gg;
            sh[0][b][hh] = (_Float16)(og * fast_tanh(c1[q]));
        }
        __syncthreads();

        // ---- layer 2: gates = h1_t @ l4Wih + h2 @ l4Whh + b4
        {
            v8f acc = {};
            #pragma unroll
            for (int ks = 0; ks < 3; ++ks)
                acc = wmma_acc(load_a_frag(&sh[0][0][0], ks, lrow, half), bWi4[ks], acc);
            acc = wmma_acc(load_a_frag(&sh[0][0][0], 3, lrow, half), expand_tail(cWi4, half), acc);
            #pragma unroll
            for (int ks = 0; ks < 3; ++ks)
                acc = wmma_acc(load_a_frag(&sh[1][0][0], ks, lrow, half), bWh4[ks], acc);
            acc = wmma_acc(load_a_frag(&sh[1][0][0], 3, lrow, half), expand_tail(cWh4, half), acc);
            #pragma unroll
            for (int r = 0; r < 8; ++r) {
                const int m = r + (half ? 8 : 0);
                sg[m][nCol] = acc[r] + b4v;
            }
        }
        __syncthreads();
        #pragma unroll
        for (int q = 0; q < 2; ++q) {
            const int p = tid + q * NTHREADS;
            const int b = p / HDIM, hh = p % HDIM;
            const float ig = fast_sigmoid(sg[b][hh]);
            const float fg = fast_sigmoid(sg[b][hh + HDIM]);
            const float gg = fast_tanh   (sg[b][hh + 2 * HDIM]);
            const float og = fast_sigmoid(sg[b][hh + 3 * HDIM]);
            c2[q] = fg * c2[q] + ig * gg;
            sh[1][b][hh] = (_Float16)(og * fast_tanh(c2[q]));
        }
        __syncthreads();

        // ---- layer 3: gates = h2_t @ l4Wih + h3 @ l4Whh + b4
        {
            v8f acc = {};
            #pragma unroll
            for (int ks = 0; ks < 3; ++ks)
                acc = wmma_acc(load_a_frag(&sh[1][0][0], ks, lrow, half), bWi4[ks], acc);
            acc = wmma_acc(load_a_frag(&sh[1][0][0], 3, lrow, half), expand_tail(cWi4, half), acc);
            #pragma unroll
            for (int ks = 0; ks < 3; ++ks)
                acc = wmma_acc(load_a_frag(&sh[2][0][0], ks, lrow, half), bWh4[ks], acc);
            acc = wmma_acc(load_a_frag(&sh[2][0][0], 3, lrow, half), expand_tail(cWh4, half), acc);
            #pragma unroll
            for (int r = 0; r < 8; ++r) {
                const int m = r + (half ? 8 : 0);
                sg[m][nCol] = acc[r] + b4v;
            }
        }
        __syncthreads();
        #pragma unroll
        for (int q = 0; q < 2; ++q) {
            const int p = tid + q * NTHREADS;
            const int b = p / HDIM, hh = p % HDIM;
            const float ig = fast_sigmoid(sg[b][hh]);
            const float fg = fast_sigmoid(sg[b][hh + HDIM]);
            const float gg = fast_tanh   (sg[b][hh + 2 * HDIM]);
            const float og = fast_sigmoid(sg[b][hh + 3 * HDIM]);
            c3[q] = fg * c3[q] + ig * gg;
            sh[2][b][hh] = (_Float16)(og * fast_tanh(c3[q]));
        }
        // top-of-loop barrier separates these writes from the next step's reads
    }
    __syncthreads();

    // ---- head: out[b] = h3_last[b,:] @ lin_w + lin_b
    if (tid < MB) {
        float s = linb[0];
        #pragma unroll 4
        for (int hh = 0; hh < HDIM; ++hh)
            s += (float)sh[2][tid][hh] * linw[hh];
        out[bBase + tid] = s;
    }
}

extern "C" void kernel_launch(void* const* d_in, const int* in_sizes, int n_in,
                              void* d_out, int out_size, void* d_ws, size_t ws_size,
                              hipStream_t stream) {
    (void)in_sizes; (void)n_in; (void)d_ws; (void)ws_size; (void)out_size;
    const float* x      = (const float*)d_in[0];
    const float* wihmu  = (const float*)d_in[1];
    const float* wihrho = (const float*)d_in[2];
    const float* epsih  = (const float*)d_in[3];
    const float* whhmu  = (const float*)d_in[4];
    const float* whhrho = (const float*)d_in[5];
    const float* epshh  = (const float*)d_in[6];
    const float* bmu    = (const float*)d_in[7];
    const float* brho   = (const float*)d_in[8];
    const float* epsb   = (const float*)d_in[9];
    const float* l4wih  = (const float*)d_in[10];
    const float* l4whh  = (const float*)d_in[11];
    const float* l4b    = (const float*)d_in[12];
    const float* linw   = (const float*)d_in[13];
    const float* linb   = (const float*)d_in[14];
    float* out = (float*)d_out;

    dim3 grid(256 / MB);       // 16 independent batch chunks
    dim3 block(NTHREADS);      // 25 waves = 25 N-tiles of the 400 gate columns
    hipLaunchKernelGGL(lstm3_fused, grid, block, 0, stream,
                       x, wihmu, wihrho, epsih, whhmu, whhrho, epshh,
                       bmu, brho, epsb, l4wih, l4whh, l4b, linw, linb, out);
}
